// GIN_37795712205240
// MI455X (gfx1250) — matrix operand order repaired and checked by reading, not compiled, not run on hardware
//
#include <hip/hip_runtime.h>
#include <hip/hip_bf16.h>
#include <math.h>

typedef __attribute__((ext_vector_type(16))) __bf16 bf16x16;
typedef __attribute__((ext_vector_type(8)))  __bf16 bf16x8;
typedef __attribute__((ext_vector_type(8)))  float  f32x8;
typedef __attribute__((ext_vector_type(4)))  float  f32x4;

// ---------------------------------------------------------------- helpers

__device__ __forceinline__ f32x8 zero8() {
    f32x8 z;
#pragma unroll
    for (int i = 0; i < 8; ++i) z[i] = 0.f;
    return z;
}

__device__ __forceinline__ f32x8 wmma_bf16(bf16x16 a, bf16x16 b, f32x8 c) {
    // D = A(16x32 bf16) * B(32x16 bf16) + C(16x16 f32)
    return __builtin_amdgcn_wmma_f32_16x16x32_bf16(
        /*neg_a=*/false, a, /*neg_b=*/false, b,
        /*c_mod=*/(short)0, c, /*reuse_a=*/false, /*reuse_b=*/false);
}

// Native fp32 global atomic add (global_atomic_add_f32, no CAS loop).
__device__ __forceinline__ void gatomic_add(float* p, float v) {
    unsafeAtomicAdd(p, v);
}

// B operand: lane (0-15) = column n, elems 0-7 = K[p..p+7], elems 8-15 = K[p+16..p+23].
__device__ __forceinline__ bf16x16 load_b16x16(const __bf16* p) {
    bf16x8 q0 = *(const bf16x8*)p;
    bf16x8 q1 = *(const bf16x8*)(p + 16);
    bf16x16 r;
#pragma unroll
    for (int i = 0; i < 8; ++i) { r[i] = q0[i]; r[i + 8] = q1[i]; }
    return r;
}

// A operand element order (ISA 16-bit A 16x32 layout):
// elems 0-7 = K[off..off+7], elems 8-15 = K[off+16..off+23]
__device__ __forceinline__ bf16x16 pack4(f32x4 q0, f32x4 q1, f32x4 q2, f32x4 q3) {
    bf16x16 r;
#pragma unroll
    for (int i = 0; i < 4; ++i) {
        r[i]      = (__bf16)q0[i];
        r[4 + i]  = (__bf16)q1[i];
        r[8 + i]  = (__bf16)q2[i];
        r[12 + i] = (__bf16)q3[i];
    }
    return r;
}

// ---------------------------------------------------------------- prep
// Transpose+convert weights to bf16 [out][in]; fold bias+BN into gamma/delta;
// sum readout biases.
__global__ void prep_kernel(
    const float* fh_W1, const float* fh_b1, const float* fh_g1, const float* fh_bt1,
    const float* fh_m1, const float* fh_v1,
    const float* fh_W2, const float* fh_b2, const float* fh_g2, const float* fh_bt2,
    const float* fh_m2, const float* fh_v2,
    const float* conv_W1, const float* conv_b1, const float* conv_g1, const float* conv_bt1,
    const float* conv_m1, const float* conv_v1,
    const float* conv_W2, const float* conv_b2, const float* conv_g2, const float* conv_bt2,
    const float* conv_m2, const float* conv_v2,
    const float* lin_b,
    __bf16* fhW1t, __bf16* fhW2t, __bf16* cW1t, __bf16* cW2t,
    float* coef, float* bias_sum)
{
    const int gsz = gridDim.x * blockDim.x;
    const int tid = blockIdx.x * blockDim.x + threadIdx.x;

    for (int i = tid; i < 64 * 128; i += gsz) {            // fh_W1 [128][64] -> [64][128]
        int n = i >> 7, k = i & 127;
        fhW1t[i] = (__bf16)fh_W1[k * 64 + n];
    }
    for (int i = tid; i < 64 * 64; i += gsz) {             // fh_W2 [64][64] -> [64][64]^T
        int n = i >> 6, k = i & 63;
        fhW2t[i] = (__bf16)fh_W2[k * 64 + n];
    }
    for (int i = tid; i < 3 * 64 * 64; i += gsz) {         // conv W1/W2 per layer
        int l = i >> 12, r = i & 4095;
        int n = r >> 6, k = r & 63;
        cW1t[i] = (__bf16)conv_W1[l * 4096 + k * 64 + n];
        cW2t[i] = (__bf16)conv_W2[l * 4096 + k * 64 + n];
    }
    // coef layout: [layer 0..3][sub 0..1][ gamma[64] | delta[64] ]
    for (int i = tid; i < 4 * 2 * 64; i += gsz) {
        int layer = i >> 7, sub = (i >> 6) & 1, j = i & 63;
        const float *g, *bt, *m, *v, *b;
        if (layer == 0) {
            if (sub == 0) { g = fh_g1; bt = fh_bt1; m = fh_m1; v = fh_v1; b = fh_b1; }
            else          { g = fh_g2; bt = fh_bt2; m = fh_m2; v = fh_v2; b = fh_b2; }
        } else {
            int l = layer - 1;
            if (sub == 0) { g = conv_g1 + l * 64; bt = conv_bt1 + l * 64; m = conv_m1 + l * 64; v = conv_v1 + l * 64; b = conv_b1 + l * 64; }
            else          { g = conv_g2 + l * 64; bt = conv_bt2 + l * 64; m = conv_m2 + l * 64; v = conv_v2 + l * 64; b = conv_b2 + l * 64; }
        }
        float s = g[j] * rsqrtf(v[j] + 1e-5f);
        coef[layer * 256 + sub * 128 + j]      = s;                         // gamma
        coef[layer * 256 + sub * 128 + 64 + j] = (b[j] - m[j]) * s + bt[j]; // delta
    }
    if (tid < 2) {
        float s = 0.f;
        for (int l = 0; l < 4; ++l) s += lin_b[l * 2 + tid];
        bias_sum[tid] = s;
    }
}

// ---------------------------------------------------------------- zero
__global__ void zero_kernel(float* __restrict__ p, int n) {
    int i = blockIdx.x * blockDim.x + threadIdx.x;
    if (i < n) p[i] = 0.f;
}

// ---------------------------------------------------------------- embed + first MLP
// One wave handles 16 nodes: gather emb rows (K=128), lin1 (16 WMMA),
// h -> LDS (bf16), lin2 (8 WMMA), store x[N,64] f32.
__global__ void __launch_bounds__(256) embed_mlp_kernel(
    const int* __restrict__ node_ids, const float* __restrict__ emb,
    const __bf16* __restrict__ W1t, const __bf16* __restrict__ W2t,
    const float* __restrict__ coef, float* __restrict__ x, int N)
{
    __shared__ __align__(32) __bf16 hb[8][16][64];
    const int tid = threadIdx.x, w = tid >> 5, lane = tid & 31;
    const int lr = lane & 15, hi = lane >> 4;
    const int base = (blockIdx.x * 8 + w) * 16;

    int nodeA = base + lr; if (nodeA >= N) nodeA = N - 1;
    int id = node_ids[nodeA];
    if (id < 0) id = 0; if (id >= N) id = N - 1;
    const float* pr = emb + (size_t)id * 128;

    bf16x16 a[4];
#pragma unroll
    for (int c = 0; c < 4; ++c) {
        int off = c * 32 + hi * 8;
        f32x4 q0 = *(const f32x4*)(pr + off);
        f32x4 q1 = *(const f32x4*)(pr + off + 4);
        f32x4 q2 = *(const f32x4*)(pr + off + 16);
        f32x4 q3 = *(const f32x4*)(pr + off + 20);
        a[c] = pack4(q0, q1, q2, q3);
    }

    f32x8 acc[4];
#pragma unroll
    for (int j = 0; j < 4; ++j) acc[j] = zero8();
#pragma unroll
    for (int j = 0; j < 4; ++j) {
        const int n = j * 16 + lr;
        const __bf16* pw = W1t + n * 128;
#pragma unroll
        for (int c = 0; c < 4; ++c)
            acc[j] = wmma_bf16(a[c], load_b16x16(pw + c * 32 + hi * 8), acc[j]);
    }
    // epilogue 1: BN+ReLU, stage to LDS as bf16
#pragma unroll
    for (int j = 0; j < 4; ++j) {
        const int n = j * 16 + lr;
        const float ga = coef[n], de = coef[64 + n];
#pragma unroll
        for (int r = 0; r < 8; ++r) {
            const int m = r + hi * 8;
            hb[w][m][n] = (__bf16)fmaxf(acc[j][r] * ga + de, 0.f);
        }
    }
    __syncthreads();

    bf16x16 a2[2];
#pragma unroll
    for (int c = 0; c < 2; ++c) {
        const int off = c * 32 + hi * 8;
        bf16x8 q0 = *(const bf16x8*)&hb[w][lr][off];
        bf16x8 q1 = *(const bf16x8*)&hb[w][lr][off + 16];
        bf16x16 av;
#pragma unroll
        for (int i = 0; i < 8; ++i) { av[i] = q0[i]; av[i + 8] = q1[i]; }
        a2[c] = av;
    }

    f32x8 acc2[4];
#pragma unroll
    for (int j = 0; j < 4; ++j) acc2[j] = zero8();
#pragma unroll
    for (int j = 0; j < 4; ++j) {
        const int n = j * 16 + lr;
        const __bf16* pw = W2t + n * 64;
#pragma unroll
        for (int c = 0; c < 2; ++c)
            acc2[j] = wmma_bf16(a2[c], load_b16x16(pw + c * 32 + hi * 8), acc2[j]);
    }
#pragma unroll
    for (int j = 0; j < 4; ++j) {
        const int n = j * 16 + lr;
        const float ga = coef[128 + n], de = coef[192 + n];
#pragma unroll
        for (int r = 0; r < 8; ++r) {
            const int m = r + hi * 8;
            const int node = base + m;
            if (node < N) x[(size_t)node * 64 + n] = fmaxf(acc2[j][r] * ga + de, 0.f);
        }
    }
}

// ---------------------------------------------------------------- GIN conv MLP
// x := MLP(x + agg) per node, in place (row-local). K=64 both linears.
__global__ void __launch_bounds__(256) conv_mlp_kernel(
    float* __restrict__ x, const float* __restrict__ agg,
    const __bf16* __restrict__ W1t, const __bf16* __restrict__ W2t,
    const float* __restrict__ coef, int N)
{
    __shared__ __align__(32) __bf16 hb[8][16][64];
    const int tid = threadIdx.x, w = tid >> 5, lane = tid & 31;
    const int lr = lane & 15, hi = lane >> 4;
    const int base = (blockIdx.x * 8 + w) * 16;

    int nodeA = base + lr; if (nodeA >= N) nodeA = N - 1;
    const float* px = x   + (size_t)nodeA * 64;
    const float* pa = agg + (size_t)nodeA * 64;

    bf16x16 a[2];
#pragma unroll
    for (int c = 0; c < 2; ++c) {
        int off = c * 32 + hi * 8;
        f32x4 q0 = *(const f32x4*)(px + off)      + *(const f32x4*)(pa + off);
        f32x4 q1 = *(const f32x4*)(px + off + 4)  + *(const f32x4*)(pa + off + 4);
        f32x4 q2 = *(const f32x4*)(px + off + 16) + *(const f32x4*)(pa + off + 16);
        f32x4 q3 = *(const f32x4*)(px + off + 20) + *(const f32x4*)(pa + off + 20);
        a[c] = pack4(q0, q1, q2, q3);
    }

    f32x8 acc[4];
#pragma unroll
    for (int j = 0; j < 4; ++j) acc[j] = zero8();
#pragma unroll
    for (int j = 0; j < 4; ++j) {
        const int n = j * 16 + lr;
        const __bf16* pw = W1t + n * 64;
#pragma unroll
        for (int c = 0; c < 2; ++c)
            acc[j] = wmma_bf16(a[c], load_b16x16(pw + c * 32 + hi * 8), acc[j]);
    }
#pragma unroll
    for (int j = 0; j < 4; ++j) {
        const int n = j * 16 + lr;
        const float ga = coef[n], de = coef[64 + n];
#pragma unroll
        for (int r = 0; r < 8; ++r) {
            const int m = r + hi * 8;
            hb[w][m][n] = (__bf16)fmaxf(acc[j][r] * ga + de, 0.f);
        }
    }
    __syncthreads();

    bf16x16 a2[2];
#pragma unroll
    for (int c = 0; c < 2; ++c) {
        const int off = c * 32 + hi * 8;
        bf16x8 q0 = *(const bf16x8*)&hb[w][lr][off];
        bf16x8 q1 = *(const bf16x8*)&hb[w][lr][off + 16];
        bf16x16 av;
#pragma unroll
        for (int i = 0; i < 8; ++i) { av[i] = q0[i]; av[i + 8] = q1[i]; }
        a2[c] = av;
    }

    f32x8 acc2[4];
#pragma unroll
    for (int j = 0; j < 4; ++j) acc2[j] = zero8();
#pragma unroll
    for (int j = 0; j < 4; ++j) {
        const int n = j * 16 + lr;
        const __bf16* pw = W2t + n * 64;
#pragma unroll
        for (int c = 0; c < 2; ++c)
            acc2[j] = wmma_bf16(a2[c], load_b16x16(pw + c * 32 + hi * 8), acc2[j]);
    }
#pragma unroll
    for (int j = 0; j < 4; ++j) {
        const int n = j * 16 + lr;
        const float ga = coef[128 + n], de = coef[192 + n];
#pragma unroll
        for (int r = 0; r < 8; ++r) {
            const int m = r + hi * 8;
            const int node = base + m;
            if (node < N) x[(size_t)node * 64 + n] = fmaxf(acc2[j][r] * ga + de, 0.f);
        }
    }
}

// ---------------------------------------------------------------- edge scatter-add
// agg[dst] += x[src]; one wave per edge, 2 channels per lane.
// Native global_atomic_add_f32 via unsafeAtomicAdd (no CAS retry loop).
__global__ void __launch_bounds__(256) edge_agg_kernel(
    const float* __restrict__ x,
    const int* __restrict__ src, const int* __restrict__ dst,
    float* __restrict__ agg, int E)
{
    const int t = blockIdx.x * blockDim.x + threadIdx.x;
    const int e = t >> 5;
    const int c = t & 31;
    if (e < E) {
        const int s = src[e], d = dst[e];
        const float v0 = x[(size_t)s * 64 + c];
        const float v1 = x[(size_t)s * 64 + c + 32];
        gatomic_add(&agg[(size_t)d * 64 + c], v0);
        gatomic_add(&agg[(size_t)d * 64 + c + 32], v1);
    }
}

// ---------------------------------------------------------------- readout
// out_acc[g][t] += sum_{node in g} (x[node] @ W[64x2]); LDS-binned (G*T=256).
__global__ void __launch_bounds__(256) readout_kernel(
    const float* __restrict__ x, const int* __restrict__ batch,
    const float* __restrict__ W, float* __restrict__ out_acc, int N)
{
    __shared__ float sW[128];
    __shared__ float sacc[256];
    const int tid = threadIdx.x;
    if (tid < 128) sW[tid] = W[tid];
    sacc[tid] = 0.f;
    __syncthreads();

    const int node = blockIdx.x * 256 + tid;
    if (node < N) {
        const float* r = x + (size_t)node * 64;
        float y0 = 0.f, y1 = 0.f;
#pragma unroll
        for (int h4 = 0; h4 < 16; ++h4) {
            f32x4 xv = *(const f32x4*)(r + h4 * 4);
#pragma unroll
            for (int i = 0; i < 4; ++i) {
                const int h = h4 * 4 + i;
                y0 += xv[i] * sW[2 * h];
                y1 += xv[i] * sW[2 * h + 1];
            }
        }
        const int g = batch[node];
        atomicAdd(&sacc[g * 2],     y0);   // ds_add_f32
        atomicAdd(&sacc[g * 2 + 1], y1);
    }
    __syncthreads();
    const float v = sacc[tid];
    if (v != 0.f) gatomic_add(&out_acc[tid], v);
}

// ---------------------------------------------------------------- softmax
__global__ void softmax_kernel(const float* __restrict__ acc, const float* __restrict__ bias_sum,
                               float* __restrict__ out, int G)
{
    const int g = blockIdx.x * blockDim.x + threadIdx.x;
    if (g < G) {
        const float a = acc[g * 2]     + bias_sum[0];
        const float b = acc[g * 2 + 1] + bias_sum[1];
        const float m = fmaxf(a, b);
        const float e0 = expf(a - m), e1 = expf(b - m);
        const float inv = 1.f / (e0 + e1);
        out[g * 2]     = e0 * inv;
        out[g * 2 + 1] = e1 * inv;
    }
}

// ---------------------------------------------------------------- launch
extern "C" void kernel_launch(void* const* d_in, const int* in_sizes, int n_in,
                              void* d_out, int out_size, void* d_ws, size_t ws_size,
                              hipStream_t stream)
{
    const int*   node_ids = (const int*)d_in[0];
    const int*   edge     = (const int*)d_in[1];
    const int*   batch    = (const int*)d_in[2];
    const float* emb      = (const float*)d_in[3];
    const float* fh_W1    = (const float*)d_in[4];
    const float* fh_b1    = (const float*)d_in[5];
    const float* fh_g1    = (const float*)d_in[6];
    const float* fh_bt1   = (const float*)d_in[7];
    const float* fh_m1    = (const float*)d_in[8];
    const float* fh_v1    = (const float*)d_in[9];
    const float* fh_W2    = (const float*)d_in[10];
    const float* fh_b2    = (const float*)d_in[11];
    const float* fh_g2    = (const float*)d_in[12];
    const float* fh_bt2   = (const float*)d_in[13];
    const float* fh_m2    = (const float*)d_in[14];
    const float* fh_v2    = (const float*)d_in[15];
    const float* conv_W1  = (const float*)d_in[16];
    const float* conv_b1  = (const float*)d_in[17];
    const float* conv_g1  = (const float*)d_in[18];
    const float* conv_bt1 = (const float*)d_in[19];
    const float* conv_m1  = (const float*)d_in[20];
    const float* conv_v1  = (const float*)d_in[21];
    const float* conv_W2  = (const float*)d_in[22];
    const float* conv_b2  = (const float*)d_in[23];
    const float* conv_g2  = (const float*)d_in[24];
    const float* conv_bt2 = (const float*)d_in[25];
    const float* conv_m2  = (const float*)d_in[26];
    const float* conv_v2  = (const float*)d_in[27];
    const float* lin_W    = (const float*)d_in[28];
    const float* lin_b    = (const float*)d_in[29];

    const int N = in_sizes[0];
    const int E = in_sizes[1] / 2;
    const int G = out_size / 2;            // T == 2

    // workspace carving (256B aligned regions)
    size_t off = 0;
    char* base = (char*)d_ws;
    auto carve = [&](size_t bytes) -> char* {
        char* p = base + off;
        off = (off + bytes + 255) & ~(size_t)255;
        return p;
    };
    float*  x        = (float*)carve((size_t)N * 64 * sizeof(float));
    float*  agg      = (float*)carve((size_t)N * 64 * sizeof(float));
    float*  out_acc  = (float*)carve((size_t)G * 2 * sizeof(float));
    float*  coef     = (float*)carve(4 * 256 * sizeof(float));
    float*  bias_sum = (float*)carve(2 * sizeof(float));
    __bf16* fhW1t    = (__bf16*)carve(64 * 128 * sizeof(__bf16));
    __bf16* fhW2t    = (__bf16*)carve(64 * 64 * sizeof(__bf16));
    __bf16* cW1t     = (__bf16*)carve(3 * 64 * 64 * sizeof(__bf16));
    __bf16* cW2t     = (__bf16*)carve(3 * 64 * 64 * sizeof(__bf16));
    (void)ws_size; (void)n_in;

    prep_kernel<<<16, 256, 0, stream>>>(
        fh_W1, fh_b1, fh_g1, fh_bt1, fh_m1, fh_v1,
        fh_W2, fh_b2, fh_g2, fh_bt2, fh_m2, fh_v2,
        conv_W1, conv_b1, conv_g1, conv_bt1, conv_m1, conv_v1,
        conv_W2, conv_b2, conv_g2, conv_bt2, conv_m2, conv_v2,
        lin_b, fhW1t, fhW2t, cW1t, cW2t, coef, bias_sum);

    zero_kernel<<<1, 256, 0, stream>>>(out_acc, G * 2);

    const int mlp_blocks  = (N + 127) / 128;         // 8 waves x 16 nodes
    const int ro_blocks   = (N + 255) / 256;
    const int zero_n      = N * 64;
    const int zero_blocks = (zero_n + 255) / 256;
    const int edge_blocks = (E * 32 + 255) / 256;

    embed_mlp_kernel<<<mlp_blocks, 256, 0, stream>>>(node_ids, emb, fhW1t, fhW2t,
                                                     coef + 0 * 256, x, N);
    readout_kernel<<<ro_blocks, 256, 0, stream>>>(x, batch, lin_W + 0 * 128, out_acc, N);

    for (int l = 0; l < 3; ++l) {
        zero_kernel<<<zero_blocks, 256, 0, stream>>>(agg, zero_n);
        edge_agg_kernel<<<edge_blocks, 256, 0, stream>>>(x, edge, edge + E, agg, E);
        conv_mlp_kernel<<<mlp_blocks, 256, 0, stream>>>(x, agg,
                                                        cW1t + l * 4096, cW2t + l * 4096,
                                                        coef + (1 + l) * 256, N);
        readout_kernel<<<ro_blocks, 256, 0, stream>>>(x, batch, lin_W + (l + 1) * 128,
                                                      out_acc, N);
    }

    softmax_kernel<<<(G + 127) / 128, 128, 0, stream>>>(out_acc, bias_sum, (float*)d_out, G);
}
